// TransformerLayer_50921132261697
// MI455X (gfx1250) — matrix operand order repaired and checked by reading, not compiled, not run on hardware
//
#include <hip/hip_runtime.h>

// CDNA5 / gfx1250, wave32. Fused Set-Transformer: one workgroup per batch item.
// All activations stay in LDS; all M=64 GEMMs use v_wmma_f32_16x16x32_f16.

typedef __attribute__((ext_vector_type(16))) _Float16 v16h;
typedef __attribute__((ext_vector_type(8)))  float    v8f;

#define DEVINL __device__ __forceinline__

struct KParams {
    const float* in[49];
    float* out;
};

// ---- WMMA fragment loaders (wave32 layouts per CDNA5 ISA 7.12.2) ----------
// A: 16x32 f16, row-major in LDS with stride lda (elements).
// lane 0-15: row m0+l, K pairs {2j,2j+1} in v0-3, {16+2j,...} in v4-7
// lane 16-31: same rows, K offset +8 (v0-3) / +24 (v4-7)
DEVINL v16h frag_a(const _Float16* A, int lda, int m0, int k0, int lane) {
    const int l = lane & 15, hi = lane >> 4;
    const _Float16* row = A + (m0 + l) * lda + k0 + hi * 8;
    union { unsigned u[8]; v16h h; } f;
#pragma unroll
    for (int j = 0; j < 4; ++j) {
        f.u[j]     = *(const unsigned*)(row + 2 * j);
        f.u[4 + j] = *(const unsigned*)(row + 16 + 2 * j);
    }
    return f.h;
}

// B: 32x16 f16 from transposed weight Wt[n][k] (row n = output col), stride K.
// lane 0-15: col n0+l, K pairs {2j,2j+1} in v0-3, {16+2j,...} in v4-7; lane 16-31: K offset +8.
DEVINL v16h frag_b(const _Float16* Wt, int ldk, int n0, int k0, int lane) {
    const int l = lane & 15, hi = lane >> 4;
    const _Float16* row = Wt + (n0 + l) * ldk + k0 + hi * 8;
    union { unsigned u[8]; v16h h; } f;
#pragma unroll
    for (int j = 0; j < 4; ++j) {
        f.u[j]     = *(const unsigned*)(row + 2 * j);
        f.u[4 + j] = *(const unsigned*)(row + 16 + 2 * j);
    }
    return f.h;
}

// GEMM: D[64,N] = A[64,K] (f16 LDS, stride lda) x W[K,N] (as Wt[N][K] f16 LDS).
// Epilogue gets (m, n, acc_f32). 8 waves cover the 4*(N/16) tiles.
template <typename Epi>
DEVINL void gemm64(const _Float16* A, int lda, const _Float16* Wt,
                   int K, int N, int tid, Epi epi) {
    const int wave = tid >> 5, lane = tid & 31;
    const int tiles = 4 * (N >> 4);
    for (int t = wave; t < tiles; t += 8) {
        const int mi = (t & 3) << 4;
        const int ni = (t >> 2) << 4;
        v8f c = {};
        for (int k = 0; k < K; k += 32) {
            v16h a = frag_a(A, lda, mi, k, lane);
            v16h b = frag_b(Wt, K, ni, k, lane);
            c = __builtin_amdgcn_wmma_f32_16x16x32_f16(
                    false, a, false, b, (short)0, c, false, false);
        }
        const int l = lane & 15, hi = lane >> 4;
#pragma unroll
        for (int r = 0; r < 8; ++r) epi(mi + hi * 8 + r, ni + l, c[r]);
    }
}

// f32 global [K,N] -> f16 LDS transposed [N][K] (B-operand staging, L2-served)
DEVINL void cvtT(const float* W, int K, int N, int nlog2, _Float16* Wt, int tid) {
    const int KN = K * N;
    for (int i = tid; i < KN; i += 256) {
        const int k = i >> nlog2, n = i & (N - 1);
        Wt[n * K + k] = (_Float16)W[i];
    }
}

// f32 global -> f16 LDS, same (row-major) layout
DEVINL void cvt(const float* W, int KN, _Float16* dst, int tid) {
    for (int i = tid; i < KN; i += 256) dst[i] = (_Float16)W[i];
}

// Small GEMM (R rows <= 4, width 128): out[r,n] = bias[n] + x[r,:K] . wh[:,n]
// resrelu: out = x[r,n] + relu(dot + bias)   (residual == x, ldx == 128)
DEVINL void small_mm(const float* x, int ldx, int R, int K,
                     const _Float16* wh, const float* bias,
                     float* out, int tid, bool resrelu) {
    const int tot = R * 128;
    for (int i = tid; i < tot; i += 256) {
        const int r = i >> 7, n = i & 127;
        const float* xr = x + r * ldx;
        float s = bias[n];
        for (int k = 0; k < K; ++k) s += xr[k] * (float)wh[k * 128 + n];
        out[i] = resrelu ? (x[r * ldx + n] + fmaxf(s, 0.f)) : s;
    }
}

#define SM_SCALE 0.08838834764831845f  // 1/sqrt(128)

// Inducing attention: q=4 rows (f32 qp[4][128]), K/V f16 [64][128].
// ob[4][128] = qp + softmax(qp.Kh/sqrt(128)) @ Vh   (per-head dh=32)
DEVINL void attn_ind(const float* qp, const _Float16* Kf, const _Float16* Vf,
                     float* sc, float* ob, int tid) {
    for (int i = tid; i < 1024; i += 256) {          // sc[(h*4+q)*64 + k]
        const int k = i & 63, q = (i >> 6) & 3, h = i >> 8;
        const float* qr = qp + q * 128 + h * 32;
        const _Float16* kr = Kf + k * 128 + h * 32;
        float s = 0.f;
        for (int d = 0; d < 32; ++d) s += qr[d] * (float)kr[d];
        sc[i] = s * SM_SCALE;
    }
    __syncthreads();
    if (tid < 16) {
        float* row = sc + tid * 64;
        float mx = row[0];
        for (int k = 1; k < 64; ++k) mx = fmaxf(mx, row[k]);
        float sum = 0.f;
        for (int k = 0; k < 64; ++k) { float e = __expf(row[k] - mx); row[k] = e; sum += e; }
        const float inv = 1.f / sum;
        for (int k = 0; k < 64; ++k) row[k] *= inv;
    }
    __syncthreads();
    for (int i = tid; i < 512; i += 256) {           // ob[q*128+c]
        const int q = i >> 7, c = i & 127, h = c >> 5;
        const float* p = sc + (h * 4 + q) * 64;
        float s = qp[i];
        for (int k = 0; k < 64; ++k) s += p[k] * (float)Vf[k * 128 + c];
        ob[i] = s;
    }
    __syncthreads();
}

// Broadcast attention: q=64 (f16 Qf[64][128]), k=4 (f32 kh/vh[4][128]) -> Of f16
DEVINL void attn_broad(const _Float16* Qf, const float* kh, const float* vh,
                       float* sc, _Float16* Of, int tid) {
    for (int i = tid; i < 1024; i += 256) {          // sc[(h*64+q)*4 + k]
        const int k = i & 3, q = (i >> 2) & 63, h = i >> 8;
        const _Float16* qr = Qf + q * 128 + h * 32;
        const float* kr = kh + k * 128 + h * 32;
        float s = 0.f;
        for (int d = 0; d < 32; ++d) s += (float)qr[d] * kr[d];
        sc[i] = s * SM_SCALE;
    }
    __syncthreads();
    {   // 256 rows of 4: one per thread
        float* row = sc + tid * 4;
        const float mx = fmaxf(fmaxf(row[0], row[1]), fmaxf(row[2], row[3]));
        const float e0 = __expf(row[0] - mx), e1 = __expf(row[1] - mx);
        const float e2 = __expf(row[2] - mx), e3 = __expf(row[3] - mx);
        const float inv = 1.f / (e0 + e1 + e2 + e3);
        row[0] = e0 * inv; row[1] = e1 * inv; row[2] = e2 * inv; row[3] = e3 * inv;
    }
    __syncthreads();
    for (int i = tid; i < 8192; i += 256) {          // Of[q*128+c]
        const int q = i >> 7, c = i & 127, h = c >> 5;
        const float* p = sc + (h * 64 + q) * 4;
        float s = (float)Qf[i];
        for (int k = 0; k < 4; ++k) s += p[k] * vh[k * 128 + c];
        Of[i] = (_Float16)s;
    }
    __syncthreads();
}

// Decoder attention: q=1 (f32 qd[128]), K/V f16 [64][128] -> ob f32 [128]
DEVINL void attn_dec(const float* qd, const _Float16* Kf, const _Float16* Vf,
                     float* sc, float* ob, int tid) {
    if (tid < 256) {                                 // sc[h*64+k]
        const int k = tid & 63, h = tid >> 6;
        const float* qr = qd + h * 32;
        const _Float16* kr = Kf + k * 128 + h * 32;
        float s = 0.f;
        for (int d = 0; d < 32; ++d) s += qr[d] * (float)kr[d];
        sc[tid] = s * SM_SCALE;
    }
    __syncthreads();
    if (tid < 4) {
        float* row = sc + tid * 64;
        float mx = row[0];
        for (int k = 1; k < 64; ++k) mx = fmaxf(mx, row[k]);
        float sum = 0.f;
        for (int k = 0; k < 64; ++k) { float e = __expf(row[k] - mx); row[k] = e; sum += e; }
        const float inv = 1.f / sum;
        for (int k = 0; k < 64; ++k) row[k] *= inv;
    }
    __syncthreads();
    if (tid < 128) {
        const int c = tid, h = c >> 5;
        const float* p = sc + h * 64;
        float s = qd[c];
        for (int k = 0; k < 64; ++k) s += p[k] * (float)Vf[k * 128 + c];
        ob[c] = s;
    }
    __syncthreads();
}

// ---------------------------------------------------------------------------
__global__ void __launch_bounds__(256, 1)
settrans_fused(KParams P) {
    const int b = blockIdx.x;
    const int tid = threadIdx.x;

    extern __shared__ char smem_raw[];
    _Float16* sX = (_Float16*)smem_raw;      // v' [64][256] (32KB); later v2/v3 [64][128]
    _Float16* sK = sX + 64 * 256;            // Kp  [64][128]
    _Float16* sV = sK + 64 * 128;            // Vp  [64][128]
    _Float16* sQ = sV + 64 * 128;            // Qp  [64][128]
    _Float16* sO = sQ + 64 * 128;            // attn out [64][128]
    _Float16* sW = sO + 64 * 128;            // weight buffer, 32768 f16 (64KB)
    float* qp = (float*)(sW + 32768);        // [4][128]
    float* ob = qp + 512;                    // [4][128]
    float* hb = ob + 512;                    // [4][128]
    float* kh = hb + 512;                    // [4][128]
    float* vh = kh + 512;                    // [4][128]
    float* sc = vh + 512;                    // 1024 scores

    const float* vIn = P.in[0] + (size_t)b * 64 * 256;
    const float* wIn = P.in[1] + (size_t)b * 64 * 64;
    const float* qIn = P.in[2] + (size_t)b * 256;

    // ---- RankAdd: v' = v + W @ pe_w + pe_b  ([64,64]x[64,256]) ----
    cvt(wIn, 64 * 64, sK, tid);                       // A tile (reuse sK)
    cvtT(P.in[3], 64, 256, 8, sW, tid);               // pe_w -> [256][64]
    __syncthreads();
    { const float* peb = P.in[4];
      gemm64(sK, 64, sW, 64, 256, tid, [&](int m, int n, float d) {
          sX[m * 256 + n] = (_Float16)(vIn[m * 256 + n] + d + peb[n]); }); }
    __syncthreads();

    // ---- ISAB1 m0: Kp/Vp = v' @ wk/wv  ([64,256]x[256,128]) ----
    cvtT(P.in[8], 256, 128, 7, sW, tid); __syncthreads();
    { const float* bk = P.in[9];
      gemm64(sX, 256, sW, 256, 128, tid, [&](int m, int n, float d) {
          sK[m * 128 + n] = (_Float16)(d + bk[n]); }); }
    __syncthreads();
    cvtT(P.in[10], 256, 128, 7, sW, tid); __syncthreads();
    { const float* bv = P.in[11];
      gemm64(sX, 256, sW, 256, 128, tid, [&](int m, int n, float d) {
          sV[m * 128 + n] = (_Float16)(d + bv[n]); }); }
    __syncthreads();
    // Qp1 = I1 @ wq + bq  (M=4: VALU)
    cvt(P.in[6], 128 * 128, sW, tid); __syncthreads();
    small_mm(P.in[5], 128, 4, 128, sW, P.in[7], qp, tid, false);
    __syncthreads();
    attn_ind(qp, sK, sV, sc, ob, tid);
    // H1 = ob + relu(ob @ wo + bo)
    cvt(P.in[12], 128 * 128, sW, tid); __syncthreads();
    small_mm(ob, 128, 4, 128, sW, P.in[13], hb, tid, true);
    __syncthreads();
    // Kh1/Vh1 = H1 @ m1_wk/m1_wv
    cvt(P.in[16], 128 * 128, sW, tid); __syncthreads();
    small_mm(hb, 128, 4, 128, sW, P.in[17], kh, tid, false);
    __syncthreads();
    cvt(P.in[18], 128 * 128, sW, tid); __syncthreads();
    small_mm(hb, 128, 4, 128, sW, P.in[19], vh, tid, false);
    __syncthreads();
    // Qp2 = v' @ m1_wq  ([64,256]x[256,128])
    cvtT(P.in[14], 256, 128, 7, sW, tid); __syncthreads();
    { const float* bq = P.in[15];
      gemm64(sX, 256, sW, 256, 128, tid, [&](int m, int n, float d) {
          sQ[m * 128 + n] = (_Float16)(d + bq[n]); }); }
    __syncthreads();
    attn_broad(sQ, kh, vh, sc, sO, tid);
    // v2 = O + relu(O @ m1_wo + bo) -> sX (v' dead now), stride 128
    cvtT(P.in[20], 128, 128, 7, sW, tid); __syncthreads();
    { const float* bo = P.in[21];
      gemm64(sO, 128, sW, 128, 128, tid, [&](int m, int n, float d) {
          sX[m * 128 + n] = (_Float16)((float)sO[m * 128 + n] + fmaxf(d + bo[n], 0.f)); }); }
    __syncthreads();

    // ---- ISAB2 (all [64,128]x[128,128]) ----
    cvtT(P.in[25], 128, 128, 7, sW, tid); __syncthreads();
    { const float* bk = P.in[26];
      gemm64(sX, 128, sW, 128, 128, tid, [&](int m, int n, float d) {
          sK[m * 128 + n] = (_Float16)(d + bk[n]); }); }
    __syncthreads();
    cvtT(P.in[27], 128, 128, 7, sW, tid); __syncthreads();
    { const float* bv = P.in[28];
      gemm64(sX, 128, sW, 128, 128, tid, [&](int m, int n, float d) {
          sV[m * 128 + n] = (_Float16)(d + bv[n]); }); }
    __syncthreads();
    cvt(P.in[23], 128 * 128, sW, tid); __syncthreads();
    small_mm(P.in[22], 128, 4, 128, sW, P.in[24], qp, tid, false);
    __syncthreads();
    attn_ind(qp, sK, sV, sc, ob, tid);
    cvt(P.in[29], 128 * 128, sW, tid); __syncthreads();
    small_mm(ob, 128, 4, 128, sW, P.in[30], hb, tid, true);
    __syncthreads();
    cvt(P.in[33], 128 * 128, sW, tid); __syncthreads();
    small_mm(hb, 128, 4, 128, sW, P.in[34], kh, tid, false);
    __syncthreads();
    cvt(P.in[35], 128 * 128, sW, tid); __syncthreads();
    small_mm(hb, 128, 4, 128, sW, P.in[36], vh, tid, false);
    __syncthreads();
    cvtT(P.in[31], 128, 128, 7, sW, tid); __syncthreads();
    { const float* bq = P.in[32];
      gemm64(sX, 128, sW, 128, 128, tid, [&](int m, int n, float d) {
          sQ[m * 128 + n] = (_Float16)(d + bq[n]); }); }
    __syncthreads();
    attn_broad(sQ, kh, vh, sc, sO, tid);
    cvtT(P.in[37], 128, 128, 7, sW, tid); __syncthreads();
    { const float* bo = P.in[38];
      gemm64(sO, 128, sW, 128, 128, tid, [&](int m, int n, float d) {
          sX[m * 128 + n] = (_Float16)((float)sO[m * 128 + n] + fmaxf(d + bo[n], 0.f)); }); }
    __syncthreads();

    // ---- Decoder MAB(Q, v3) ----
    cvtT(P.in[41], 128, 128, 7, sW, tid); __syncthreads();
    { const float* bk = P.in[42];
      gemm64(sX, 128, sW, 128, 128, tid, [&](int m, int n, float d) {
          sK[m * 128 + n] = (_Float16)(d + bk[n]); }); }
    __syncthreads();
    cvtT(P.in[43], 128, 128, 7, sW, tid); __syncthreads();
    { const float* bv = P.in[44];
      gemm64(sX, 128, sW, 128, 128, tid, [&](int m, int n, float d) {
          sV[m * 128 + n] = (_Float16)(d + bv[n]); }); }
    __syncthreads();
    // Qd = Q[b] @ dec_wq + bq  (1x256 x 256x128)
    cvt(P.in[39], 256 * 128, sW, tid); __syncthreads();
    small_mm(qIn, 256, 1, 256, sW, P.in[40], qp, tid, false);
    __syncthreads();
    attn_dec(qp, sK, sV, sc, ob, tid);
    // o = Od + relu(Od @ dec_wo + bo)
    cvt(P.in[45], 128 * 128, sW, tid); __syncthreads();
    small_mm(ob, 128, 1, 128, sW, P.in[46], hb, tid, true);
    __syncthreads();
    // out = o @ out_w + out_b  (1x128 x 128x256) -> global f32
    cvt(P.in[47], 128 * 256, sW, tid); __syncthreads();
    {
        const float* outb = P.in[48];
        float s = outb[tid];
        for (int k = 0; k < 128; ++k) s += hb[k] * (float)sW[k * 256 + tid];
        P.out[(size_t)b * 256 + tid] = s;
    }
}

extern "C" void kernel_launch(void* const* d_in, const int* in_sizes, int n_in,
                              void* d_out, int out_size, void* d_ws, size_t ws_size,
                              hipStream_t stream) {
    (void)out_size; (void)d_ws; (void)ws_size;
    KParams P;
    for (int i = 0; i < 49 && i < n_in; ++i) P.in[i] = (const float*)d_in[i];
    P.out = (float*)d_out;
    const int B = in_sizes[0] / (64 * 256);   // 4096
    // LDS: f16 {v' 16384, K/V/Q/O 4*8192, W 32768} + f32 {5*512 + 1024}
    const size_t smem = (size_t)(64 * 256 + 4 * 64 * 128 + 32768) * sizeof(_Float16)
                      + (size_t)(5 * 512 + 1024) * sizeof(float);
    settrans_fused<<<B, 256, smem, stream>>>(P);
}
